// MSTHH_82755429859467
// MI455X (gfx1250) — matrix operand order repaired and checked by reading, not compiled, not run on hardware
//
#include <hip/hip_runtime.h>
#include <hip/hip_bf16.h>

#define B_ 8
#define T_ 128
#define N_ 256
#define D_ 64
#define LN_EPS 1e-5f

typedef _Float16 v16h __attribute__((ext_vector_type(16)));
typedef float    v8f  __attribute__((ext_vector_type(8)));

// D = A(16x32 f16) * B(32x16 f16) + C(16x16 f32)
__device__ __forceinline__ v8f wmma16(v16h a, v16h b, v8f c) {
  return __builtin_amdgcn_wmma_f32_16x16x32_f16(false, a, false, b, (short)0, c,
                                                false, false);
}

// A-fragment: 16x32 tile of row-major f16 matrix (LDS or global), leading dim ld.
// Lane l (m = l&15, h = l>>4): elems 0..7 -> K = h*8+e ; elems 8..15 -> K = 16+h*8+e.
__device__ __forceinline__ v16h ld_a(const _Float16* s, int row0, int k0, int ld, int lane) {
  const int m = lane & 15, h = lane >> 4;
  const _Float16* p = s + (size_t)(row0 + m) * ld + k0 + h * 8;
  v16h a;
#pragma unroll
  for (int e = 0; e < 8; ++e) a[e] = p[e];
#pragma unroll
  for (int e = 0; e < 8; ++e) a[8 + e] = p[16 + e];
  return a;
}

// Same, but source is f32 (global), converted on the fly.
__device__ __forceinline__ v16h ld_a_f32(const float* s, int row0, int k0, int ld, int lane) {
  const int m = lane & 15, h = lane >> 4;
  const float* p = s + (size_t)(row0 + m) * ld + k0 + h * 8;
  v16h a;
#pragma unroll
  for (int e = 0; e < 8; ++e) a[e] = (_Float16)p[e];
#pragma unroll
  for (int e = 0; e < 8; ++e) a[8 + e] = (_Float16)p[16 + e];
  return a;
}

// B-fragment: B[k][n] taken from X with X[n][k] row-major, leading dim ld
// (i.e. the operand is stored "transposed": weight matrices W[e][d], K[j][d], V^T[d][j]).
// Lane l (n = n0 + (l&15), h = l>>4): elems 0..7 -> K = k0+h*8+e ; elems 8..15 -> K = k0+16+h*8+e.
__device__ __forceinline__ v16h ld_bt(const _Float16* x, int n0, int k0, int ld, int lane) {
  const int n = (lane & 15) + n0, h = lane >> 4;
  const _Float16* p = x + (size_t)n * ld + k0 + h * 8;
  v16h b;
#pragma unroll
  for (int e = 0; e < 8; ++e) b[e] = p[e];
#pragma unroll
  for (int e = 0; e < 8; ++e) b[8 + e] = p[16 + e];
  return b;
}

// ---------------------------------------------------------------------------
// adp = softmax_n( relu(e1 @ e2) )  -> f16 [N][N]
// one block per row m; 256 threads, one column each.
// ---------------------------------------------------------------------------
__global__ void __launch_bounds__(256) k_adp(const float* __restrict__ e1,
                                             const float* __restrict__ e2,
                                             _Float16* __restrict__ adp) {
  __shared__ float sE1[32];
  __shared__ float sRed[256];
  const int m = blockIdx.x, j = threadIdx.x;
  if (j < 32) sE1[j] = e1[m * 32 + j];
  __syncthreads();
  float s = 0.f;
#pragma unroll
  for (int k = 0; k < 32; ++k) s += sE1[k] * e2[k * N_ + j];
  s = fmaxf(s, 0.f);
  // row max
  sRed[j] = s;
  __syncthreads();
  for (int off = 128; off; off >>= 1) {
    if (j < off) sRed[j] = fmaxf(sRed[j], sRed[j + off]);
    __syncthreads();
  }
  const float mx = sRed[0];
  __syncthreads();
  const float p = expf(s - mx);
  sRed[j] = p;
  __syncthreads();
  for (int off = 128; off; off >>= 1) {
    if (j < off) sRed[j] += sRed[j + off];
    __syncthreads();
  }
  const float inv = 1.0f / sRed[0];
  adp[(size_t)m * N_ + j] = (_Float16)(p * inv);
}

// ---------------------------------------------------------------------------
// Fused QKV projection + per-node temporal attention.  One block per (b,n).
// out = softmax(Q K^T/(8 tau) + log(decay)) V   -> aligned[B,T,N,D] f32
// ---------------------------------------------------------------------------
__global__ void __launch_bounds__(256) k_qkv_attn(
    const float* __restrict__ Hi, const float* __restrict__ Hj,
    const float* __restrict__ Wq, const float* __restrict__ bq,
    const float* __restrict__ Wk, const float* __restrict__ bk,
    const float* __restrict__ Wv, const float* __restrict__ bv,
    const float* __restrict__ lgam, const float* __restrict__ ltau,
    float* __restrict__ out) {
  __shared__ _Float16 sH[T_ * D_];       // Hj -> Hi -> Q  (aliased phases)
  __shared__ _Float16 sW[3 * D_ * D_];   // Wq | Wk | Wv
  __shared__ _Float16 sK[T_ * D_];       // K[t][d]
  __shared__ _Float16 sVt[D_ * T_];      // V^T[d][t]
  __shared__ _Float16 sP[T_ * T_];       // attn[i][j]

  const int b = blockIdx.x / N_;
  const int n = blockIdx.x % N_;
  const int tid = threadIdx.x;
  const int lane = tid & 31, wave = tid >> 5;
  const int l15 = lane & 15, lh = lane >> 4;
  const int rb = wave * 16;  // this wave's 16 time-rows

  for (int i = tid; i < D_ * D_; i += 256) {
    sW[i]               = (_Float16)Wq[i];
    sW[D_ * D_ + i]     = (_Float16)Wk[i];
    sW[2 * D_ * D_ + i] = (_Float16)Wv[i];
  }
  for (int i = tid; i < T_ * D_; i += 256) {
    const int t = i >> 6, d = i & 63;
    sH[i] = (_Float16)Hj[(((size_t)b * T_ + t) * N_ + n) * D_ + d];
  }
  __syncthreads();

  // ---- K = Hj Wk^T + bk -> sK ----
  {
    v8f acc[4] = {};
#pragma unroll
    for (int kk = 0; kk < 2; ++kk) {
      v16h a = ld_a(sH, rb, kk * 32, D_, lane);
#pragma unroll
      for (int et = 0; et < 4; ++et)
        acc[et] = wmma16(a, ld_bt(sW + D_ * D_, et * 16, kk * 32, D_, lane), acc[et]);
    }
#pragma unroll
    for (int et = 0; et < 4; ++et) {
      const int e = et * 16 + l15;
      const float be = bk[e];
#pragma unroll
      for (int r = 0; r < 8; ++r)
        sK[(rb + r + 8 * lh) * D_ + e] = (_Float16)(acc[et][r] + be);
    }
  }
  // ---- V = Hj Wv^T + bv -> sVt (transposed) ----
  {
    v8f acc[4] = {};
#pragma unroll
    for (int kk = 0; kk < 2; ++kk) {
      v16h a = ld_a(sH, rb, kk * 32, D_, lane);
#pragma unroll
      for (int et = 0; et < 4; ++et)
        acc[et] = wmma16(a, ld_bt(sW + 2 * D_ * D_, et * 16, kk * 32, D_, lane), acc[et]);
    }
#pragma unroll
    for (int et = 0; et < 4; ++et) {
      const int e = et * 16 + l15;
      const float be = bv[e];
#pragma unroll
      for (int r = 0; r < 8; ++r)
        sVt[e * T_ + (rb + r + 8 * lh)] = (_Float16)(acc[et][r] + be);
    }
  }
  __syncthreads();
  // Hi replaces Hj
  for (int i = tid; i < T_ * D_; i += 256) {
    const int t = i >> 6, d = i & 63;
    sH[i] = (_Float16)Hi[(((size_t)b * T_ + t) * N_ + n) * D_ + d];
  }
  __syncthreads();
  // ---- Q = Hi Wq^T + bq (regs), then overwrite sH with Q ----
  v8f accQ[4] = {};
#pragma unroll
  for (int kk = 0; kk < 2; ++kk) {
    v16h a = ld_a(sH, rb, kk * 32, D_, lane);
#pragma unroll
    for (int et = 0; et < 4; ++et)
      accQ[et] = wmma16(a, ld_bt(sW, et * 16, kk * 32, D_, lane), accQ[et]);
  }
  __syncthreads();  // everyone done reading Hi
#pragma unroll
  for (int et = 0; et < 4; ++et) {
    const int e = et * 16 + l15;
    const float be = bq[e];
#pragma unroll
    for (int r = 0; r < 8; ++r)
      sH[(rb + r + 8 * lh) * D_ + e] = (_Float16)(accQ[et][r] + be);
  }
  __syncthreads();

  // ---- S = Q K^T * scale + log(decay); row softmax -> sP (f16) ----
  const float gamma = fmaxf(expf(lgam[0]), 0.01f);
  const float tau   = fmaxf(expf(ltau[0]), 0.01f);
  const float scale = 1.0f / (8.0f * tau);
  const float inv127 = 1.0f / 127.0f;

  v8f accS[8] = {};
#pragma unroll
  for (int kk = 0; kk < 2; ++kk) {
    v16h a = ld_a(sH, rb, kk * 32, D_, lane);
#pragma unroll
    for (int jt = 0; jt < 8; ++jt)
      accS[jt] = wmma16(a, ld_bt(sK, jt * 16, kk * 32, D_, lane), accS[jt]);
  }
#pragma unroll
  for (int r = 0; r < 8; ++r) {
    const int i = rb + r + 8 * lh;
    const float ti = i * inv127;
    float mx = -1e30f;
#pragma unroll
    for (int jt = 0; jt < 8; ++jt) {
      const int j = jt * 16 + l15;
      const float tj = j * inv127;
      float s = accS[jt][r] * scale + logf(expf(-gamma * fabsf(ti - tj)) + 1e-8f);
      accS[jt][r] = s;
      mx = fmaxf(mx, s);
    }
    for (int off = 8; off; off >>= 1) mx = fmaxf(mx, __shfl_xor(mx, off, 16));
    float sum = 0.f;
#pragma unroll
    for (int jt = 0; jt < 8; ++jt) {
      const float p = expf(accS[jt][r] - mx);
      accS[jt][r] = p;
      sum += p;
    }
    for (int off = 8; off; off >>= 1) sum += __shfl_xor(sum, off, 16);
    const float inv = 1.0f / sum;
#pragma unroll
    for (int jt = 0; jt < 8; ++jt)
      sP[(size_t)i * T_ + jt * 16 + l15] = (_Float16)(accS[jt][r] * inv);
  }
  __syncthreads();

  // ---- O = P V -> aligned ----
  v8f accO[4] = {};
#pragma unroll
  for (int kk = 0; kk < 4; ++kk) {
    v16h a = ld_a(sP, rb, kk * 32, T_, lane);
#pragma unroll
    for (int dt = 0; dt < 4; ++dt)
      accO[dt] = wmma16(a, ld_bt(sVt, dt * 16, kk * 32, T_, lane), accO[dt]);
  }
#pragma unroll
  for (int dt = 0; dt < 4; ++dt) {
    const int d = dt * 16 + l15;
#pragma unroll
    for (int r = 0; r < 8; ++r) {
      const int i = rb + r + 8 * lh;
      out[(((size_t)b * T_ + i) * N_ + n) * D_ + d] = accO[dt][r];
    }
  }
}

// ---------------------------------------------------------------------------
// GCN: h = relu((adp @ x) Wg^T + bg);  x1 = LN1(x + h).  One block per (b,t).
// ---------------------------------------------------------------------------
__global__ void __launch_bounds__(256) k_gcn(
    const float* __restrict__ x, const _Float16* __restrict__ adp,
    const float* __restrict__ Wg, const float* __restrict__ bg,
    const float* __restrict__ g1, const float* __restrict__ b1,
    float* __restrict__ x1) {
  __shared__ _Float16 sXt[D_ * N_];  // x^T [d][n]
  __shared__ _Float16 sY[N_ * D_];   // y [m][d]
  __shared__ _Float16 sWg[D_ * D_];
  __shared__ float    sF[N_ * D_];   // h (f32)

  const int tid = threadIdx.x, lane = tid & 31, wave = tid >> 5;
  const int l15 = lane & 15, lh = lane >> 4;
  const size_t base = (size_t)blockIdx.x * N_ * D_;

  for (int i = tid; i < N_ * D_; i += 256) {
    const int nn = i >> 6, d = i & 63;
    sXt[d * N_ + nn] = (_Float16)x[base + i];
  }
  for (int i = tid; i < D_ * D_; i += 256) sWg[i] = (_Float16)Wg[i];
  __syncthreads();

  // y = adp @ x  (M=256, K=256, Ncol=64); adp A-tiles gathered straight from L2
#pragma unroll
  for (int mt = 0; mt < 2; ++mt) {
    const int m0 = (wave * 2 + mt) * 16;
    v8f acc[4] = {};
    for (int kk = 0; kk < 8; ++kk) {
      v16h a = ld_a(adp, m0, kk * 32, N_, lane);
#pragma unroll
      for (int dt = 0; dt < 4; ++dt)
        acc[dt] = wmma16(a, ld_bt(sXt, dt * 16, kk * 32, N_, lane), acc[dt]);
    }
#pragma unroll
    for (int dt = 0; dt < 4; ++dt) {
      const int d = dt * 16 + l15;
#pragma unroll
      for (int r = 0; r < 8; ++r)
        sY[(m0 + r + 8 * lh) * D_ + d] = (_Float16)acc[dt][r];
    }
  }
  __syncthreads();

  // h = relu(y Wg^T + bg) -> sF
#pragma unroll
  for (int mt = 0; mt < 2; ++mt) {
    const int m0 = (wave * 2 + mt) * 16;
    v8f acc[4] = {};
#pragma unroll
    for (int kk = 0; kk < 2; ++kk) {
      v16h a = ld_a(sY, m0, kk * 32, D_, lane);
#pragma unroll
      for (int et = 0; et < 4; ++et)
        acc[et] = wmma16(a, ld_bt(sWg, et * 16, kk * 32, D_, lane), acc[et]);
    }
#pragma unroll
    for (int et = 0; et < 4; ++et) {
      const int e = et * 16 + l15;
      const float be = bg[e];
#pragma unroll
      for (int r = 0; r < 8; ++r)
        sF[(m0 + r + 8 * lh) * D_ + e] = fmaxf(acc[et][r] + be, 0.f);
    }
  }
  __syncthreads();

  // residual + LN1, one row per thread
  {
    const int m = tid;
    const float* xr = x + base + (size_t)m * D_;
    float s = 0.f, ss = 0.f;
    for (int d = 0; d < D_; ++d) {
      const float v = xr[d] + sF[m * D_ + d];
      s += v; ss += v * v;
    }
    const float mean = s * (1.0f / D_);
    const float var = ss * (1.0f / D_) - mean * mean;
    const float inv = rsqrtf(var + LN_EPS);
    for (int d = 0; d < D_; ++d) {
      const float v = xr[d] + sF[m * D_ + d];
      x1[base + (size_t)m * D_ + d] = (v - mean) * inv * g1[d] + b1[d];
    }
  }
}

// ---------------------------------------------------------------------------
// Fused GRU scan + LN2 + final residual. 16 blocks, each owns 128 (b,n) rows
// (b = blk>>1, n in [128*(blk&1), +128)), loops over all T internally.
// out[b,t,n,:] = x1[b,t,n,:] + LN2(x1[b,t,n,:] + h_t[n,:])
// ---------------------------------------------------------------------------
__global__ void __launch_bounds__(256) k_gru(
    const float* __restrict__ x1,
    const float* __restrict__ Wih, const float* __restrict__ Whh,
    const float* __restrict__ bih, const float* __restrict__ bhh,
    const float* __restrict__ g2, const float* __restrict__ b2,
    float* __restrict__ out) {
  __shared__ _Float16 sWih[3 * D_ * D_];
  __shared__ _Float16 sWhh[3 * D_ * D_];
  __shared__ float    sHs[128 * D_];   // h (f32)
  __shared__ _Float16 sHh[128 * D_];   // h (f16 for WMMA A)
  __shared__ _Float16 sGi[128 * 192];  // gi (f16, pre-bias)

  const int b = blockIdx.x >> 1;
  const int n0 = (blockIdx.x & 1) * 128;
  const int tid = threadIdx.x, lane = tid & 31, wave = tid >> 5;
  const int l15 = lane & 15, lh = lane >> 4;
  const int rb = wave * 16;

  for (int i = tid; i < 3 * D_ * D_; i += 256) {
    sWih[i] = (_Float16)Wih[i];
    sWhh[i] = (_Float16)Whh[i];
  }
  for (int i = tid; i < 128 * D_; i += 256) {
    sHs[i] = 0.f;
    sHh[i] = (_Float16)0.f;
  }
  __syncthreads();

  for (int t = 0; t < T_; ++t) {
    const float* xt = x1 + (((size_t)b * T_ + t) * N_ + n0) * D_;  // [128 x 64] contiguous

    // gi = x_t Wih^T  (M=128, K=64, N=192) -> sGi
    {
      v8f acc[12] = {};
#pragma unroll
      for (int kk = 0; kk < 2; ++kk) {
        v16h a = ld_a_f32(xt, rb, kk * 32, D_, lane);
#pragma unroll
        for (int ct = 0; ct < 12; ++ct)
          acc[ct] = wmma16(a, ld_bt(sWih, ct * 16, kk * 32, D_, lane), acc[ct]);
      }
#pragma unroll
      for (int ct = 0; ct < 12; ++ct) {
        const int c = ct * 16 + l15;
#pragma unroll
        for (int r = 0; r < 8; ++r)
          sGi[(rb + r + 8 * lh) * 192 + c] = (_Float16)acc[ct][r];
      }
    }
    __syncthreads();

    // gh = h Whh^T ; gates; update h
    {
      v8f acc[12] = {};
#pragma unroll
      for (int kk = 0; kk < 2; ++kk) {
        v16h a = ld_a(sHh, rb, kk * 32, D_, lane);
#pragma unroll
        for (int ct = 0; ct < 12; ++ct)
          acc[ct] = wmma16(a, ld_bt(sWhh, ct * 16, kk * 32, D_, lane), acc[ct]);
      }
#pragma unroll
      for (int c4 = 0; c4 < 4; ++c4) {
        const int dd = c4 * 16 + l15;
        const float bir = bih[dd], biz = bih[64 + dd], bin_ = bih[128 + dd];
        const float bhr = bhh[dd], bhz = bhh[64 + dd], bhn = bhh[128 + dd];
#pragma unroll
        for (int r = 0; r < 8; ++r) {
          const int m = rb + r + 8 * lh;
          const float ir  = (float)sGi[m * 192 + dd] + bir;
          const float iz  = (float)sGi[m * 192 + 64 + dd] + biz;
          const float in_ = (float)sGi[m * 192 + 128 + dd] + bin_;
          const float hr = acc[c4][r] + bhr;
          const float hz = acc[c4 + 4][r] + bhz;
          const float hn = acc[c4 + 8][r] + bhn;
          const float rr = 1.0f / (1.0f + expf(-(ir + hr)));
          const float zz = 1.0f / (1.0f + expf(-(iz + hz)));
          const float nn = tanhf(in_ + rr * hn);
          const float hv = (1.0f - zz) * nn + zz * sHs[m * D_ + dd];
          sHs[m * D_ + dd] = hv;
          sHh[m * D_ + dd] = (_Float16)hv;
        }
      }
    }
    __syncthreads();

    // LN2 + residual, one row per thread (128 rows)
    if (tid < 128) {
      const int m = tid;
      const float* xr = xt + (size_t)m * D_;
      float s = 0.f, ss = 0.f;
      for (int d = 0; d < D_; ++d) {
        const float v = xr[d] + sHs[m * D_ + d];
        s += v; ss += v * v;
      }
      const float mean = s * (1.0f / D_);
      const float var = ss * (1.0f / D_) - mean * mean;
      const float inv = rsqrtf(var + LN_EPS);
      float* orow = out + (((size_t)b * T_ + t) * N_ + n0 + m) * D_;
      for (int d = 0; d < D_; ++d) {
        const float v = xr[d] + sHs[m * D_ + d];
        orow[d] = xr[d] + ((v - mean) * inv * g2[d] + b2[d]);
      }
    }
    __syncthreads();
  }
}

extern "C" void kernel_launch(void* const* d_in, const int* in_sizes, int n_in,
                              void* d_out, int out_size, void* d_ws, size_t ws_size,
                              hipStream_t stream) {
  (void)in_sizes; (void)n_in; (void)out_size; (void)ws_size;
  const float* H_i = (const float*)d_in[0];
  const float* H_j = (const float*)d_in[1];
  const float* Wq  = (const float*)d_in[2];  const float* bq = (const float*)d_in[3];
  const float* Wk  = (const float*)d_in[4];  const float* bk = (const float*)d_in[5];
  const float* Wv  = (const float*)d_in[6];  const float* bv = (const float*)d_in[7];
  const float* lg  = (const float*)d_in[8];  const float* lt = (const float*)d_in[9];
  const float* e1  = (const float*)d_in[10]; const float* e2 = (const float*)d_in[11];
  const float* Wg  = (const float*)d_in[12]; const float* bg = (const float*)d_in[13];
  const float* g1  = (const float*)d_in[14]; const float* b1 = (const float*)d_in[15];
  const float* Wih = (const float*)d_in[16]; const float* Whh = (const float*)d_in[17];
  const float* bih = (const float*)d_in[18]; const float* bhh = (const float*)d_in[19];
  const float* g2  = (const float*)d_in[20]; const float* b2 = (const float*)d_in[21];
  float* outp = (float*)d_out;

  const size_t ELEMS = (size_t)B_ * T_ * N_ * D_;
  float* aligned = (float*)d_ws;
  float* x1 = aligned + ELEMS;
  _Float16* adp = (_Float16*)(x1 + ELEMS);

  k_adp<<<N_, 256, 0, stream>>>(e1, e2, adp);
  k_qkv_attn<<<B_ * N_, 256, 0, stream>>>(H_i, H_j, Wq, bq, Wk, bk, Wv, bv, lg, lt,
                                          aligned);
  k_gcn<<<B_ * T_, 256, 0, stream>>>(aligned, adp, Wg, bg, g1, b1, x1);
  k_gru<<<B_ * 2, 256, 0, stream>>>(x1, Wih, Whh, bih, bhh, g2, b2, outp);
}
